// VoxelModule_46170898432069
// MI455X (gfx1250) — compile-verified, matrix operand order checked by reading, not run on hardware
//
#include <hip/hip_runtime.h>

typedef __attribute__((ext_vector_type(16))) _Float16 v16h;
typedef __attribute__((ext_vector_type(8)))  float    v8f;

#define BATCH   8
#define NPTS    100000
#define VOX     40
#define NBINS   (VOX * VOX * VOX)            // 64000 compact bins
#define TOTPTS  (BATCH * NPTS)               // 800000
#define ORDER_OFF 0
#define SH_OFF    (BATCH * NPTS)             // 800000
#define NB_OFF    (2 * BATCH * NPTS)         // 1600000
#define NB_ELEMS  (BATCH * NBINS * 27 * 3)   // 41472000
#define MASK_OFF  (NB_OFF + NB_ELEMS)        // 43072000

// ---------------------------------------------------------------- init scratch
__global__ void k_init(int* __restrict__ counts, int* __restrict__ maxbin) {
    int t = blockIdx.x * blockDim.x + threadIdx.x;
    if (t < BATCH * NBINS) counts[t] = 0;
    if (t < BATCH)         maxbin[t] = 0;
}

// ------------------------------------------- hash via WMMA matvec + histogram
// Each wave handles 16 consecutive points.  A(16x32 f16): row m = coords of
// point base+m at K=0..2 (lanes 0-15, VGPR0/1 per ISA layout), rest zero.
// B(32x16 f16): w = [10000,100,1] replicated in K=0..2 (lanes 0-15).
// D(16x16 f32): column n lives in lane n, so lane 0 holds h0..h7 and
// lane 16 holds h8..h15 in the 8 accumulator VGPRs.
__global__ void k_hash(const float* __restrict__ pc, int* __restrict__ hashes,
                       int* __restrict__ counts, int* __restrict__ maxbin) {
    const int lane = threadIdx.x & 31;
    const int wave = threadIdx.x >> 5;
    const int base = (blockIdx.x * 8 + wave) * 16;   // first point of this wave

    v16h a = {};
    if (lane < 16) {
        const float* p = pc + (size_t)(base + lane) * 3;
        int cx = (int)(p[0] * (VOX - 1));
        int cy = (int)(p[1] * (VOX - 1));
        int cz = (int)(p[2] * (VOX - 1));
        a[0] = (_Float16)(float)cx;
        a[1] = (_Float16)(float)cy;
        a[2] = (_Float16)(float)cz;
    }
    v16h b = {};
    if (lane < 16) {
        b[0] = (_Float16)10000.0f;
        b[1] = (_Float16)100.0f;
        b[2] = (_Float16)1.0f;
    }
    v8f c = {};
    // EXEC is all-ones here (full 256-thread block, no early exits).
    c = __builtin_amdgcn_wmma_f32_16x16x32_f16(
        /*neg_a=*/false, a, /*neg_b=*/false, b,
        /*c_mod=*/(short)0, c, /*reuse_a=*/false, /*reuse_b=*/false);
    __builtin_amdgcn_s_wait_tensorcnt(0);  // CDNA5 split-counter path (no-op)

    if (lane == 0 || lane == 16) {
        const int mofs = (lane == 16) ? 8 : 0;
        for (int j = 0; j < 8; ++j) {
            int pt = base + mofs + j;
            int h  = (int)(c[j] + 0.5f);             // exact f32 sum -> int
            hashes[pt] = h;
            int x = h / 10000, r = h % 10000, y = r / 100, z = r % 100;
            int bin  = (x * VOX + y) * VOX + z;      // compact, hash-monotone
            int bidx = pt / NPTS;
            atomicAdd(&counts[bidx * NBINS + bin], 1);
            atomicMax(&maxbin[bidx], bin);
        }
    }
}

// -------------------------------------- per-batch exclusive scan over 64k bins
__global__ void k_scan(const int* __restrict__ counts, int* __restrict__ offs) {
    const int b   = blockIdx.x;
    const int tid = threadIdx.x;                 // blockDim == 512
    __shared__ int buf[512];
    __shared__ int carry;
    if (tid == 0) carry = 0;
    __syncthreads();
    for (int chunk = 0; chunk < NBINS; chunk += 512) {   // 125 chunks
        int v = counts[b * NBINS + chunk + tid];
        buf[tid] = v;
        __syncthreads();
        for (int off = 1; off < 512; off <<= 1) {        // Hillis-Steele
            int t = (tid >= off) ? buf[tid - off] : 0;
            __syncthreads();
            buf[tid] += t;
            __syncthreads();
        }
        int incl = buf[tid];
        offs[b * NBINS + chunk + tid] = carry + incl - v;  // exclusive
        __syncthreads();
        if (tid == 511) carry += incl;                     // chunk total
        __syncthreads();
    }
}

// ------------------------------------------- scatter: order + sorted hashes
__global__ void k_scatter(const int* __restrict__ hashes, int* __restrict__ offs,
                          float* __restrict__ out) {
    int j = blockIdx.x * blockDim.x + threadIdx.x;
    if (j >= TOTPTS) return;
    int bi = j / NPTS, i = j % NPTS;
    int h = hashes[j];
    int x = h / 10000, r = h % 10000, y = r / 100, z = r % 100;
    int bin = (x * VOX + y) * VOX + z;
    int pos = atomicAdd(&offs[bi * NBINS + bin], 1);
    out[ORDER_OFF + bi * NPTS + pos] = (float)i;   // same-key positions share h
    out[SH_OFF    + bi * NPTS + pos] = (float)h;
}

// ------------------------------------------------------------------- mask
// mask = occupied && (bin != max occupied bin)  [max-hash group never emits a
// boundary at its voxel; scatter-max with 0 elsewhere is a no-op]
__global__ void k_mask(const int* __restrict__ counts, const int* __restrict__ maxbin,
                       float* __restrict__ out) {
    int t = blockIdx.x * blockDim.x + threadIdx.x;
    if (t >= BATCH * NBINS) return;
    int b = t / NBINS, v = t % NBINS;
    out[MASK_OFF + t] = (counts[t] > 0 && v != maxbin[b]) ? 1.0f : 0.0f;
}

// -------------------------------- neighbour table: constant, float4 b128 stores
__global__ void k_nb(float* __restrict__ out) {
    long long t = (long long)blockIdx.x * blockDim.x + threadIdx.x;
    long long e0 = t * 4;
    if (e0 >= (long long)NB_ELEMS) return;
    float4 w;
    float* wp = &w.x;
#pragma unroll
    for (int k = 0; k < 4; ++k) {
        int e = (int)e0 + k;
        int cc = e % 3; int q = e / 3;
        int m  = q % 27; q /= 27;
        int z  = q % VOX; q /= VOX;
        int y  = q % VOX; q /= VOX;
        int x  = q % VOX;                         // remaining q/VOX = batch
        int coord = (cc == 0) ? x : (cc == 1) ? y : z;
        int d     = (cc == 0) ? (m / 9) : (cc == 1) ? ((m / 3) % 3) : (m % 3);
        wp[k] = (float)(coord + d - 1);
    }
    *(float4*)(out + NB_OFF + e0) = w;            // NB_OFF % 4 == 0 -> 16B aligned
}

extern "C" void kernel_launch(void* const* d_in, const int* in_sizes, int n_in,
                              void* d_out, int out_size, void* d_ws, size_t ws_size,
                              hipStream_t stream) {
    const float* pc  = (const float*)d_in[0];     // (8,100000,3) f32
    float* out       = (float*)d_out;
    int* counts = (int*)d_ws;                     // 8*64000 ints
    int* offs   = counts + BATCH * NBINS;         // 8*64000 ints
    int* hashes = offs   + BATCH * NBINS;         // 800000 ints
    int* maxbin = hashes + TOTPTS;                // 8 ints   (~7.3 MB total)

    k_init   <<<(BATCH * NBINS + 255) / 256, 256, 0, stream>>>(counts, maxbin);
    k_hash   <<<TOTPTS / 128, 256, 0, stream>>>(pc, hashes, counts, maxbin);
    k_scan   <<<BATCH, 512, 0, stream>>>(counts, offs);
    k_scatter<<<(TOTPTS + 255) / 256, 256, 0, stream>>>(hashes, offs, out);
    k_mask   <<<(BATCH * NBINS + 255) / 256, 256, 0, stream>>>(counts, maxbin, out);
    k_nb     <<<(NB_ELEMS / 4 + 255) / 256, 256, 0, stream>>>(out);
}
